// Encoder_87917980549691
// MI455X (gfx1250) — compile-verified, matrix operand order checked by reading, not run on hardware
//
#include <hip/hip_runtime.h>
#include <hip/hip_bf16.h>

typedef __attribute__((ext_vector_type(16))) __bf16 v16bf;
typedef __attribute__((ext_vector_type(8)))  __bf16 v8bf;
typedef __attribute__((ext_vector_type(4)))  __bf16 v4bf;
typedef __attribute__((ext_vector_type(8)))  float  v8f;

__device__ __forceinline__ __bf16 f2bf(float f) {
    unsigned u = __builtin_bit_cast(unsigned, f);
    unsigned r = (u + 0x7FFFu + ((u >> 16) & 1u)) >> 16;   // round-to-nearest-even
    return __builtin_bit_cast(__bf16, (unsigned short)r);
}

// ---------------------------------------------------------------------------
// f32 -> bf16 conversion, 4 elems/thread (done ONCE per buffer, outside GEMM)
// ---------------------------------------------------------------------------
__global__ __launch_bounds__(256) void cvt_bf16_kernel(const float* __restrict__ in,
                                                       __bf16* __restrict__ out, int n4) {
    int i = blockIdx.x * blockDim.x + threadIdx.x;
    if (i >= n4) return;
    float4 v = reinterpret_cast<const float4*>(in)[i];
    v4bf o;
    o[0] = f2bf(v.x); o[1] = f2bf(v.y); o[2] = f2bf(v.z); o[3] = f2bf(v.w);
    reinterpret_cast<v4bf*>(out)[i] = o;
}

// ---------------------------------------------------------------------------
// Degree histogram: deg[dst[e]] += 1
// ---------------------------------------------------------------------------
__global__ __launch_bounds__(256) void degree_kernel(const int* __restrict__ dst,
                                                     float* __restrict__ deg, int E) {
    int e = blockIdx.x * blockDim.x + threadIdx.x;
    if (e < E) atomicAdd(&deg[dst[e]], 1.0f);
}

// ---------------------------------------------------------------------------
// WMMA GEMM (bf16 in, f32 out): y = x@Wa^T (+bias if !DUAL), z = x@Wb^T+bias.
// x: [N,K] bf16, Wa/Wb: [64,K] bf16.  One wave = 16 rows x 64 cols, A reused
// across 8 v_wmma per K-chunk.  Fragments are loaded as two 16-byte bf16
// vectors per lane (no conversion VALU in the hot loop).
// ---------------------------------------------------------------------------
template <bool DUAL>
__global__ __launch_bounds__(256) void gemm_wmma(
    const __bf16* __restrict__ x, const __bf16* __restrict__ Wa,
    const __bf16* __restrict__ Wb, const float* __restrict__ bias,
    float* __restrict__ y, float* __restrict__ z, int N, int K)
{
    const int lane = threadIdx.x & 31;
    const int wave = (blockIdx.x * blockDim.x + threadIdx.x) >> 5;
    const int row0 = wave << 4;
    if (row0 >= N) return;

    const int lcol = lane & 15;          // row (A) / col (B,D) within tile
    const int koff = (lane >> 4) << 3;   // 0 or 8: K sub-offset per half-wave

    float bsc[4];
#pragma unroll
    for (int t = 0; t < 4; ++t) bsc[t] = bias ? bias[t * 16 + lcol] : 0.0f;

    v8f accA[4], accB[4];
#pragma unroll
    for (int t = 0; t < 4; ++t)
#pragma unroll
        for (int g = 0; g < 8; ++g) {
            accA[t][g] = DUAL ? 0.0f : bsc[t];
            accB[t][g] = bsc[t];
        }

    const __bf16* xr = x + (size_t)(row0 + lcol) * K + koff;

    for (int kc = 0; kc < K; kc += 32) {
        const v8bf a0 = *reinterpret_cast<const v8bf*>(xr + kc);
        const v8bf a1 = *reinterpret_cast<const v8bf*>(xr + kc + 16);
        v16bf a;
#pragma unroll
        for (int j = 0; j < 8; ++j) { a[j] = a0[j]; a[j + 8] = a1[j]; }

#pragma unroll
        for (int t = 0; t < 4; ++t) {
            const __bf16* war = Wa + (size_t)(t * 16 + lcol) * K + koff;
            const v8bf wa0 = *reinterpret_cast<const v8bf*>(war + kc);
            const v8bf wa1 = *reinterpret_cast<const v8bf*>(war + kc + 16);
            v16bf ba;
#pragma unroll
            for (int j = 0; j < 8; ++j) { ba[j] = wa0[j]; ba[j + 8] = wa1[j]; }
            accA[t] = __builtin_amdgcn_wmma_f32_16x16x32_bf16(
                false, a, false, ba, (short)0, accA[t], false, false);
            if (DUAL) {
                const __bf16* wbr = Wb + (size_t)(t * 16 + lcol) * K + koff;
                const v8bf wb0 = *reinterpret_cast<const v8bf*>(wbr + kc);
                const v8bf wb1 = *reinterpret_cast<const v8bf*>(wbr + kc + 16);
                v16bf bb;
#pragma unroll
                for (int j = 0; j < 8; ++j) { bb[j] = wb0[j]; bb[j + 8] = wb1[j]; }
                accB[t] = __builtin_amdgcn_wmma_f32_16x16x32_bf16(
                    false, a, false, bb, (short)0, accB[t], false, false);
            }
        }
    }

    const int mo = (lane >> 4) << 3;   // lanes 16-31 hold rows g+8
#pragma unroll
    for (int t = 0; t < 4; ++t)
#pragma unroll
        for (int g = 0; g < 8; ++g) {
            const size_t o = (size_t)(row0 + g + mo) * 64 + t * 16 + lcol;
            y[o] = accA[t][g];
            if (DUAL) z[o] = accB[t][g];
        }
}

// ---------------------------------------------------------------------------
// Edge scatter: s[dst[e], :] += y[src[e], :]   (64 ch, 16 thr/edge x float4)
// ---------------------------------------------------------------------------
__global__ __launch_bounds__(256) void scatter_add(
    const int* __restrict__ src, const int* __restrict__ dst,
    const float* __restrict__ y, float* __restrict__ s, int E)
{
    const int t = blockIdx.x * blockDim.x + threadIdx.x;
    const int e = t >> 4;
    if (e >= E) return;
    const int c  = (t & 15) << 2;
    const int sn = src[e], dn = dst[e];
    const float4 v = *reinterpret_cast<const float4*>(y + (size_t)sn * 64 + c);
    float* sp = s + (size_t)dn * 64 + c;
    atomicAdd(sp + 0, v.x);
    atomicAdd(sp + 1, v.y);
    atomicAdd(sp + 2, v.z);
    atomicAdd(sp + 3, v.w);
}

// ---------------------------------------------------------------------------
// h = act(s / max(deg,1) + z), emitted as bf16 (next consumer is a WMMA GEMM)
// act 0 = relu, 1 = tanh
// ---------------------------------------------------------------------------
__global__ __launch_bounds__(256) void combine_kernel(
    const float* __restrict__ s, const float* __restrict__ z,
    const float* __restrict__ deg, __bf16* __restrict__ h, int n64, int act)
{
    int i = blockIdx.x * blockDim.x + threadIdx.x;
    if (i >= n64) return;
    float d = fmaxf(deg[i >> 6], 1.0f);
    float v = s[i] / d + z[i];
    v = (act == 0) ? fmaxf(v, 0.0f) : tanhf(v);
    h[i] = f2bf(v);
}

// ---------------------------------------------------------------------------
// Row-wise softmax over 64 channels
// ---------------------------------------------------------------------------
__global__ __launch_bounds__(256) void softmax_kernel(
    const float* __restrict__ logits, float* __restrict__ out, int N)
{
    int r = blockIdx.x * blockDim.x + threadIdx.x;
    if (r >= N) return;
    const float* lr = logits + (size_t)r * 64;
    float v[64];
    float m = -3.4e38f;
#pragma unroll
    for (int c = 0; c < 64; ++c) { v[c] = lr[c]; m = fmaxf(m, v[c]); }
    float sum = 0.0f;
#pragma unroll
    for (int c = 0; c < 64; ++c) { v[c] = __expf(v[c] - m); sum += v[c]; }
    const float inv = 1.0f / sum;
    float* orow = out + (size_t)r * 64;
#pragma unroll
    for (int c = 0; c < 64; ++c) orow[c] = v[c] * inv;
}

// ---------------------------------------------------------------------------
extern "C" void kernel_launch(void* const* d_in, const int* in_sizes, int n_in,
                              void* d_out, int out_size, void* d_ws, size_t ws_size,
                              hipStream_t stream)
{
    (void)n_in; (void)out_size; (void)ws_size;
    const float* x    = (const float*)d_in[0];
    const int*   ei   = (const int*)  d_in[1];
    const float* W1l  = (const float*)d_in[2];
    const float* b1l  = (const float*)d_in[3];
    const float* W1r  = (const float*)d_in[4];
    const float* W2l  = (const float*)d_in[5];
    const float* b2l  = (const float*)d_in[6];
    const float* W2r  = (const float*)d_in[7];
    const float* Wlin = (const float*)d_in[8];
    const float* blin = (const float*)d_in[9];

    const int N = in_sizes[0] / 128;     // 100000
    const int E = in_sizes[1] / 2;       // 3200000
    const int* src = ei;
    const int* dst = ei + E;

    // ---- workspace layout -------------------------------------------------
    char* ws = (char*)d_ws;
    size_t o = 0;
    auto take = [&](size_t bytes) { void* p = ws + o; o = (o + bytes + 255) & ~(size_t)255; return p; };
    float*  deg = (float*) take((size_t)N * sizeof(float));
    __bf16* xb  = (__bf16*)take((size_t)N * 128 * sizeof(__bf16));   // bf16 features
    __bf16* wb  = (__bf16*)take((size_t)32768 * sizeof(__bf16));     // bf16 weights
    __bf16* hb  = (__bf16*)take((size_t)N * 64 * sizeof(__bf16));    // h1 then h2
    float*  B0  = (float*) take((size_t)N * 64 * sizeof(float));     // y / logits
    float*  B1  = (float*) take((size_t)N * 64 * sizeof(float));     // z
    float*  B2  = (float*) take((size_t)N * 64 * sizeof(float));     // scatter accum

    __bf16* w1l = wb;                    // 64*128
    __bf16* w1r = wb + 8192;             // 64*128
    __bf16* w2l = wb + 16384;            // 64*64
    __bf16* w2r = wb + 20480;            // 64*64
    __bf16* wln = wb + 24576;            // 64*64

    const size_t nb = (size_t)N * 64 * sizeof(float);
    hipMemsetAsync(deg, 0, (size_t)N * sizeof(float), stream);
    hipMemsetAsync(B2, 0, nb, stream);

    // ---- one-time conversions to bf16 ------------------------------------
    const int xn4 = N * 128 / 4;
    cvt_bf16_kernel<<<(xn4 + 255) / 256, 256, 0, stream>>>(x, xb, xn4);
    cvt_bf16_kernel<<<8, 256, 0, stream>>>(W1l, w1l, 8192 / 4);
    cvt_bf16_kernel<<<8, 256, 0, stream>>>(W1r, w1r, 8192 / 4);
    cvt_bf16_kernel<<<4, 256, 0, stream>>>(W2l, w2l, 4096 / 4);
    cvt_bf16_kernel<<<4, 256, 0, stream>>>(W2r, w2r, 4096 / 4);
    cvt_bf16_kernel<<<4, 256, 0, stream>>>(Wlin, wln, 4096 / 4);

    const int n64     = N * 64;
    const int dgrid   = (E + 255) / 256;
    const int gblocks = ((N + 15) / 16 + 7) / 8;                 // 8 waves/block
    const int sgrid   = (int)(((size_t)E * 16 + 255) / 256);
    const int cgrid   = (n64 + 255) / 256;

    degree_kernel<<<dgrid, 256, 0, stream>>>(dst, deg, E);

    // ---- Layer 1 (project-before-aggregate: mean is linear) --------------
    gemm_wmma<true><<<gblocks, 256, 0, stream>>>(xb, w1l, w1r, b1l, B0, B1, N, 128);
    scatter_add<<<sgrid, 256, 0, stream>>>(src, dst, B0, B2, E);
    combine_kernel<<<cgrid, 256, 0, stream>>>(B2, B1, deg, hb, n64, 0);    // h1 (bf16)

    // ---- Layer 2 ---------------------------------------------------------
    hipMemsetAsync(B2, 0, nb, stream);                                     // s2 = 0
    gemm_wmma<true><<<gblocks, 256, 0, stream>>>(hb, w2l, w2r, b2l, B0, B1, N, 64);
    scatter_add<<<sgrid, 256, 0, stream>>>(src, dst, B0, B2, E);
    combine_kernel<<<cgrid, 256, 0, stream>>>(B2, B1, deg, hb, n64, 1);    // h2 (bf16)

    // ---- Classifier + softmax --------------------------------------------
    gemm_wmma<false><<<gblocks, 256, 0, stream>>>(hb, wln, nullptr, blin, B0, nullptr, N, 64);
    softmax_kernel<<<(N + 255) / 256, 256, 0, stream>>>(B0, (float*)d_out, N);
}
